// Transformer_21964462752076
// MI455X (gfx1250) — matrix operand order repaired and checked by reading, not compiled, not run on hardware
//
#include <hip/hip_runtime.h>
#include <math.h>

typedef unsigned short u16;
typedef __attribute__((ext_vector_type(16))) __bf16 v16bf;
typedef __attribute__((ext_vector_type(8)))  float  v8f;

union Frag { uint4 u[2]; v16bf v; };
static_assert(sizeof(Frag) == 32, "frag size");

#if defined(__has_builtin)
#if __has_builtin(__builtin_amdgcn_tensor_load_to_lds)
#define USE_TDM 1
#endif
#endif
#ifndef USE_TDM
#define USE_TDM 0
#endif

__device__ __forceinline__ u16 f2bf(float f) {
    unsigned u = __float_as_uint(f);
    unsigned r = (u + 0x7FFFu + ((u >> 16) & 1u)) >> 16;
    return (u16)r;
}

__device__ __forceinline__ void store4(float* p, float a, float b, float c, float d) {
    *(float4*)p = make_float4(a, b, c, d);
}
__device__ __forceinline__ void store4(u16* p, float a, float b, float c, float d) {
    *(ushort4*)p = make_ushort4(f2bf(a), f2bf(b), f2bf(c), f2bf(d));
}

#if USE_TDM
typedef __attribute__((ext_vector_type(4))) unsigned int tdm_v4u;
typedef __attribute__((ext_vector_type(8))) int tdm_v8i;
typedef __attribute__((ext_vector_type(4))) int tdm_v4i;

// TDM: DMA one 32x128 bf16 tile (row = 64B, tensor row stride = 1024 elems)
// into LDS, padding 16B after every 64B row -> LDS row stride 80B
// (matches the bank-conflict-free [128][40] u16 layout).
__device__ __forceinline__ void tdm_load_tile(const u16* gsrc, unsigned ldsaddr) {
    unsigned long long ga = (unsigned long long)(const void*)gsrc;
    tdm_v4u g0;
    g0.x = 1u;                                  // count=1 (valid user D#)
    g0.y = ldsaddr;                             // LDS byte address of tile
    g0.z = (unsigned)ga;                        // global_addr[31:0]
    g0.w = (unsigned)(ga >> 32) | (2u << 30);   // global_addr[56:32] | type=2
    tdm_v8i g1;
    // workgroup_mask=0 | data_size=1(2B) | pad_enable | pad_interval=3(16dw=64B)
    // | pad_amount=3(4dw=16B)
    g1[0] = (int)((1u << 16) | (1u << 20) | (3u << 22) | (3u << 25));
    g1[1] = (int)(1024u << 16);                       // tensor_dim0 = 1024 (lo16)
    g1[2] = (int)((1024u >> 16) | (8192u << 16));     // dim0 hi | tensor_dim1 lo
    g1[3] = (int)((8192u >> 16) | (32u << 16));       // dim1 hi | tile_dim0 = 32
    g1[4] = (int)128;                                 // tile_dim1=128, tile_dim2=0
    g1[5] = (int)1024;                                // tensor_dim0_stride lo32
    g1[6] = 0;                                        // stride hi | dim1_stride lo
    g1[7] = 0;
    tdm_v4i gz = {0, 0, 0, 0};
#if defined(__clang_major__) && (__clang_major__ >= 23)
    tdm_v8i gz8 = {0, 0, 0, 0, 0, 0, 0, 0};
    __builtin_amdgcn_tensor_load_to_lds(g0, g1, gz, gz, gz8, 0);
#else
    __builtin_amdgcn_tensor_load_to_lds(g0, g1, gz, gz, 0);
#endif
}
#endif  // USE_TDM

// ---------------------------------------------------------------- init
__global__ void init_kernel(unsigned* amax) {
    if (threadIdx.x < 32) amax[threadIdx.x] = 0u;
}

// ------------------------------------------------- weight binarize (13x 1Mx)
__global__ __launch_bounds__(256) void binarize_kernel(
    const float* __restrict__ blW, const float* __restrict__ f1W,
    const float* __restrict__ f2W, u16* __restrict__ Wb) {
    size_t i = (size_t)blockIdx.x * 256 + threadIdx.x;
    const size_t WSZ = 1024u * 1024u;
    size_t slot = i / WSZ, r = i % WSZ;
    float w = (slot == 0) ? blW[r]
            : (slot <= 6) ? f1W[(slot - 1) * WSZ + r]
                          : f2W[(slot - 7) * WSZ + r];
    Wb[i] = (w > 0.f) ? (u16)0x3F80 : (w < 0.f ? (u16)0xBF80 : (u16)0);
}

// ------------------------------------------------- Frobenius norms (13 mats)
__global__ __launch_bounds__(256) void wnorm_kernel(
    const float* __restrict__ blW, const float* __restrict__ f1W,
    const float* __restrict__ f2W, float* __restrict__ wn) {
    const int slot = blockIdx.x, tid = threadIdx.x;
    const size_t WSZ = 1024u * 1024u;
    const float* src = (slot == 0) ? blW
                     : (slot <= 6) ? f1W + (size_t)(slot - 1) * WSZ
                                   : f2W + (size_t)(slot - 7) * WSZ;
    float s = 0.f;
    for (size_t i = tid; i < WSZ; i += 256) { float w = src[i]; s += w * w; }
#pragma unroll
    for (int o = 16; o; o >>= 1) s += __shfl_xor(s, o);
    __shared__ float red[8];
    if ((tid & 31) == 0) red[tid >> 5] = s;
    __syncthreads();
    if (tid == 0) {
        float t = 0.f;
#pragma unroll
        for (int w = 0; w < 8; ++w) t += red[w];
        wn[slot] = sqrtf(t);
    }
}

// ------------------------------------------------- LayerNorm (row = 1024)
template <typename OutT, bool ADD>
__global__ __launch_bounds__(256) void ln_kernel(
    const float* __restrict__ x, const float* __restrict__ xadd,
    const float* __restrict__ g, const float* __restrict__ b,
    OutT* __restrict__ out) {
    const int row = blockIdx.x, tid = threadIdx.x;
    float4 v = ((const float4*)(x + (size_t)row * 1024))[tid];
    if (ADD) {
        float4 a = ((const float4*)(xadd + (size_t)row * 1024))[tid];
        v.x += a.x; v.y += a.y; v.z += a.z; v.w += a.w;
    }
    float s  = v.x + v.y + v.z + v.w;
    float s2 = v.x * v.x + v.y * v.y + v.z * v.z + v.w * v.w;
#pragma unroll
    for (int o = 16; o; o >>= 1) { s += __shfl_xor(s, o); s2 += __shfl_xor(s2, o); }
    __shared__ float red[2][8];
    const int lane = tid & 31, wid = tid >> 5;
    if (lane == 0) { red[0][wid] = s; red[1][wid] = s2; }
    __syncthreads();
    s = 0.f; s2 = 0.f;
#pragma unroll
    for (int w = 0; w < 8; ++w) { s += red[0][w]; s2 += red[1][w]; }
    const float mu = s * (1.f / 1024.f);
    const float var = s2 * (1.f / 1024.f) - mu * mu;
    const float rs = rsqrtf(var + 1e-5f);
    float4 gv = ((const float4*)g)[tid];
    float4 bv = ((const float4*)b)[tid];
    store4(out + (size_t)row * 1024 + tid * 4,
           (v.x - mu) * rs * gv.x + bv.x, (v.y - mu) * rs * gv.y + bv.y,
           (v.z - mu) * rs * gv.z + bv.z, (v.w - mu) * rs * gv.w + bv.w);
}

// ------------------------------------------------- dequant (+gelu/+residual)
template <bool GELU, bool ADD, typename OutT>
__global__ __launch_bounds__(256) void dequant_kernel(
    const float* __restrict__ Y, const unsigned* __restrict__ amax_bits,
    const float* __restrict__ wn, int widx,
    const float* __restrict__ xadd, OutT* __restrict__ out) {
    size_t i = ((size_t)blockIdx.x * 256 + threadIdx.x) * 4;
    const float amax = __uint_as_float(*amax_bits);
    const float scale = 127.f / amax;
    const float inv = amax * (1.f / 127.f);
    const float resc = wn[widx] * 32.f;  // ||W||_F * sqrt(1024)
    float4 y = *(const float4*)(Y + i);
    float r[4] = {y.x, y.y, y.z, y.w};
#pragma unroll
    for (int e = 0; e < 4; ++e) {
        float d = rintf(r[e] * scale) * inv * resc;
        if (GELU) d = 0.5f * d * (1.f + erff(d * 0.70710678118f));
        r[e] = d;
    }
    if (ADD) {
        float4 a = *(const float4*)(xadd + i);
        r[0] += a.x; r[1] += a.y; r[2] += a.z; r[3] += a.w;
    }
    store4(out + i, r[0], r[1], r[2], r[3]);
}

// ------------------------------------------------- BitLinear GEMM
// Y[M,1024] = A_bf16[M,1024] x W_bf16[1024,1024]^T + bias ; absmax(|Y|)
// 128x128 tile / workgroup, 8 waves 4x2, each wave 32x64 (2x4 WMMA tiles).
// Staging: TDM tensor_load_to_lds, double-buffered, overlapped via TENSORcnt.
__global__ __launch_bounds__(256) void gemm_bitlinear_kernel(
    const u16* __restrict__ A, const u16* __restrict__ W,
    const float* __restrict__ bias, float* __restrict__ Y,
    unsigned* __restrict__ amax) {
    const int tid = threadIdx.x;
    const int lane = tid & 31, wid = tid >> 5;
    const int ln = lane & 15, hi = lane >> 4;
    const int wrow = wid >> 1, wcol = wid & 1;
    const int m0 = blockIdx.y * 128;
    const int n0 = blockIdx.x * 128;
    const int K = 1024, N = 1024;

    __shared__ u16 Ash[2][128][40];  // 32 halfs + 16B pad / row (80B stride)
    __shared__ u16 Bsh[2][128][40];

    v8f acc[2][4] = {};

#if USE_TDM
    const unsigned aAddr[2] = {(unsigned)(size_t)&Ash[0][0][0],
                               (unsigned)(size_t)&Ash[1][0][0]};
    const unsigned bAddr[2] = {(unsigned)(size_t)&Bsh[0][0][0],
                               (unsigned)(size_t)&Bsh[1][0][0]};
    if (wid == 0) {  // prologue: DMA first A/B tile pair
        tdm_load_tile(&A[(size_t)m0 * K], aAddr[0]);
        tdm_load_tile(&W[(size_t)n0 * K], bAddr[0]);
    }
#endif

    int cur = 0;
    for (int kk = 0; kk < K; kk += 32) {
        __syncthreads();  // prior-iteration reads of buf[cur^1] are done
#if USE_TDM
        if (wid == 0) {
            if (kk + 32 < K) {
                // issue next tile pair, then wait for the *older* pair only:
                // DMA of tile k+1 overlaps the WMMAs on tile k
                tdm_load_tile(&A[(size_t)m0 * K + kk + 32], aAddr[cur ^ 1]);
                tdm_load_tile(&W[(size_t)n0 * K + kk + 32], bAddr[cur ^ 1]);
                __builtin_amdgcn_s_wait_tensorcnt(2);
            } else {
                __builtin_amdgcn_s_wait_tensorcnt(0);
            }
        }
#else
#pragma unroll
        for (int c = 0; c < 2; ++c) {
            int idx = tid + c * 256;
            int r = idx >> 2, sx = idx & 3;
            const u16* ag = &A[(size_t)(m0 + r) * K + kk + sx * 8];
            const u16* bg = &W[(size_t)(n0 + r) * K + kk + sx * 8];
            *(uint4*)&Ash[cur][r][sx * 8] = *(const uint4*)ag;
            *(uint4*)&Bsh[cur][r][sx * 8] = *(const uint4*)bg;
            if (kk + 32 < K) {
                __builtin_prefetch(ag + 32, 0, 1);
                __builtin_prefetch(bg + 32, 0, 1);
            }
        }
#endif
        __syncthreads();  // buf[cur] visible to all waves

        Frag af[2], bfr[4];
#pragma unroll
        for (int mt = 0; mt < 2; ++mt) {
            int r = wrow * 32 + mt * 16 + ln;
            af[mt].u[0] = *(const uint4*)&Ash[cur][r][hi * 8];
            af[mt].u[1] = *(const uint4*)&Ash[cur][r][hi * 8 + 16];
        }
#pragma unroll
        for (int nt = 0; nt < 4; ++nt) {
            int r = wcol * 64 + nt * 16 + ln;
            bfr[nt].u[0] = *(const uint4*)&Bsh[cur][r][hi * 8];
            bfr[nt].u[1] = *(const uint4*)&Bsh[cur][r][hi * 8 + 16];
        }
#pragma unroll
        for (int mt = 0; mt < 2; ++mt)
#pragma unroll
            for (int nt = 0; nt < 4; ++nt)
                acc[mt][nt] = __builtin_amdgcn_wmma_f32_16x16x32_bf16(
                    false, af[mt].v, false, bfr[nt].v, (short)0, acc[mt][nt],
                    false, false);
        cur ^= 1;
    }

    float am = 0.f;
#pragma unroll
    for (int mt = 0; mt < 2; ++mt) {
#pragma unroll
        for (int nt = 0; nt < 4; ++nt) {
            int col = n0 + wcol * 64 + nt * 16 + ln;
            float bb = bias[col];
#pragma unroll
            for (int v = 0; v < 8; ++v) {
                int row = m0 + wrow * 32 + mt * 16 + v + hi * 8;
                float val = acc[mt][nt][v] + bb;
                Y[(size_t)row * N + col] = val;
                am = fmaxf(am, fabsf(val));
            }
        }
    }
#pragma unroll
    for (int o = 16; o; o >>= 1) am = fmaxf(am, __shfl_xor(am, o));
    if (lane == 0) atomicMax(amax, __float_as_uint(am));
}

// ------------------------------------------------- Flash attention
// q=k=v = qkv bf16 [B,N,1024]; head h = cols h*64..h*64+63; scale 1/8
// block = 4 waves, 64 query rows (16/wave); KV streamed in blocks of 64
__global__ __launch_bounds__(128) void flash_attn_kernel(
    const u16* __restrict__ qkv, float* __restrict__ outp) {
    const int tid = threadIdx.x;
    const int lane = tid & 31, wid = tid >> 5;
    const int ln = lane & 15, hi = lane >> 4;
    const int b = blockIdx.y >> 4, h = blockIdx.y & 15;
    const int q0 = blockIdx.x * 64;
    const size_t RS = 1024;

    __shared__ u16 Vt[64][72];      // V transposed: [d][key]  (16B-aligned rows)
    __shared__ u16 Pb[4][16][72];   // per-wave P tile (C-layout -> A-layout)

    const u16* base = qkv + ((size_t)b * 2048) * RS + h * 64;

    // Q fragments, A-layout, kept in registers for entire KV loop
    Frag qf[2];
    {
        const u16* qrow = base + (size_t)(q0 + wid * 16 + ln) * RS;
#pragma unroll
        for (int ks = 0; ks < 2; ++ks) {
            int d0 = ks * 32 + hi * 8;
            qf[ks].u[0] = *(const uint4*)(qrow + d0);
            qf[ks].u[1] = *(const uint4*)(qrow + d0 + 16);
        }
    }

    v8f Oc[4] = {};
    float mrow[8], lrow[8];
#pragma unroll
    for (int v = 0; v < 8; ++v) { mrow[v] = -1e30f; lrow[v] = 0.f; }

    for (int jb = 0; jb < 2048; jb += 64) {
        __syncthreads();
        {   // cooperative transposed load of the V block (64 keys x 64 d)
            int j = tid >> 1;
            int d0 = (tid & 1) * 32;
            union { uint4 q[4]; u16 hh[32]; } t;
            const u16* vrow = base + (size_t)(jb + j) * RS + d0;
            const uint4* src = (const uint4*)vrow;
            t.q[0] = src[0]; t.q[1] = src[1]; t.q[2] = src[2]; t.q[3] = src[3];
#pragma unroll
            for (int e = 0; e < 32; ++e) Vt[d0 + e][j] = t.hh[e];
            if (jb + 64 < 2048) __builtin_prefetch(vrow + 64 * RS, 0, 1);
        }
        __syncthreads();

        // S = (Q K^T) — K fragments straight from global (row-major = B layout)
        v8f S[4] = {};
#pragma unroll
        for (int nt = 0; nt < 4; ++nt) {
            const u16* krow = base + (size_t)(jb + nt * 16 + ln) * RS;
#pragma unroll
            for (int ks = 0; ks < 2; ++ks) {
                Frag kf;
                int d0 = ks * 32 + hi * 8;
                kf.u[0] = *(const uint4*)(krow + d0);
                kf.u[1] = *(const uint4*)(krow + d0 + 16);
                S[nt] = __builtin_amdgcn_wmma_f32_16x16x32_bf16(
                    false, qf[ks].v, false, kf.v, (short)0, S[nt], false, false);
            }
        }

        // online softmax per query row (row = v + hi*8 within wave tile)
#pragma unroll
        for (int v = 0; v < 8; ++v) {
            float sv[4];
#pragma unroll
            for (int nt = 0; nt < 4; ++nt) sv[nt] = S[nt][v] * 0.125f;
            float sm = fmaxf(fmaxf(sv[0], sv[1]), fmaxf(sv[2], sv[3]));
#pragma unroll
            for (int o = 8; o; o >>= 1) sm = fmaxf(sm, __shfl_xor(sm, o));
            float mn = fmaxf(mrow[v], sm);
            float alpha = __expf(mrow[v] - mn);
            mrow[v] = mn;
            float ps = 0.f;
#pragma unroll
            for (int nt = 0; nt < 4; ++nt) {
                float p = __expf(sv[nt] - mn);
                ps += p;
                Pb[wid][v + hi * 8][nt * 16 + ln] = f2bf(p);
            }
#pragma unroll
            for (int o = 8; o; o >>= 1) ps += __shfl_xor(ps, o);
            lrow[v] = lrow[v] * alpha + ps;
#pragma unroll
            for (int td = 0; td < 4; ++td) Oc[td][v] *= alpha;
        }
        __syncthreads();

        // O += P x V
#pragma unroll
        for (int ks = 0; ks < 2; ++ks) {
            Frag pf;
            int k0 = ks * 32 + hi * 8;
            pf.u[0] = *(const uint4*)&Pb[wid][ln][k0];
            pf.u[1] = *(const uint4*)&Pb[wid][ln][k0 + 16];
#pragma unroll
            for (int td = 0; td < 4; ++td) {
                Frag vf;
                int j0 = ks * 32 + hi * 8;
                vf.u[0] = *(const uint4*)&Vt[td * 16 + ln][j0];
                vf.u[1] = *(const uint4*)&Vt[td * 16 + ln][j0 + 16];
                Oc[td] = __builtin_amdgcn_wmma_f32_16x16x32_bf16(
                    false, pf.v, false, vf.v, (short)0, Oc[td], false, false);
            }
        }
    }

    float* orow = outp + ((size_t)b * 2048 + q0 + wid * 16) * 1024 + h * 64;
#pragma unroll
    for (int td = 0; td < 4; ++td)
#pragma unroll
        for (int v = 0; v < 8; ++v)
            orow[(size_t)(v + hi * 8) * 1024 + td * 16 + ln] = Oc[td][v] / lrow[v];
}

// ================================================================ launch
extern "C" void kernel_launch(void* const* d_in, const int* in_sizes, int n_in,
                              void* d_out, int out_size, void* d_ws, size_t ws_size,
                              hipStream_t stream) {
    const float* x      = (const float*)d_in[0];
    const float* blg    = (const float*)d_in[1];
    const float* blb    = (const float*)d_in[2];
    const float* blW    = (const float*)d_in[3];
    const float* blbias = (const float*)d_in[4];
    const float* ffg    = (const float*)d_in[5];
    const float* ffb    = (const float*)d_in[6];
    const float* f1g    = (const float*)d_in[7];
    const float* f1b    = (const float*)d_in[8];
    const float* f1W    = (const float*)d_in[9];
    const float* f1bb   = (const float*)d_in[10];
    const float* f2g    = (const float*)d_in[11];
    const float* f2b    = (const float*)d_in[12];
    const float* f2W    = (const float*)d_in[13];
    const float* f2bb   = (const float*)d_in[14];
    const float* ng     = (const float*)d_in[15];
    const float* nb     = (const float*)d_in[16];
    (void)in_sizes; (void)n_in; (void)out_size; (void)ws_size;

    const size_t WSZ = 1024u * 1024u;  // elements per weight matrix
    const size_t MT  = 8192;           // B*N tokens

    char* ws = (char*)d_ws;
    u16* Wb = (u16*)ws;                     ws += 13 * WSZ * sizeof(u16);   // 26MB
    float*    wn   = (float*)ws;
    unsigned* amax = (unsigned*)(ws + 128); ws += 1024;
    float* X  = (float*)ws;                 ws += MT * 1024 * sizeof(float); // 32MB
    u16*   Ab = (u16*)ws;                   ws += MT * 1024 * sizeof(u16);   // 16MB (LN out / qkv)
    float* Yb = (float*)ws;                 ws += MT * 1024 * sizeof(float); // 32MB
    float* Tb = (float*)ws;                                                  // 32MB

    hipMemcpyAsync(X, x, MT * 1024 * sizeof(float), hipMemcpyDeviceToDevice, stream);
    init_kernel<<<1, 32, 0, stream>>>(amax);
    binarize_kernel<<<13 * 4096, 256, 0, stream>>>(blW, f1W, f2W, Wb);
    wnorm_kernel<<<13, 256, 0, stream>>>(blW, f1W, f2W, wn);

    const dim3 ggrid(8, 64);   // N/128, M/128
    const dim3 fgrid(32, 64);  // query blocks, B*H
    const int LNB = 8192, DQB = 8192;

    for (int i = 0; i < 6; ++i) {
        unsigned* s0 = amax + i * 4 + 0;
        unsigned* s1 = amax + i * 4 + 1;
        unsigned* s2 = amax + i * 4 + 2;
        unsigned* s3 = amax + i * 4 + 3;

        // qkv = bitlinear(x) ; attention
        ln_kernel<u16, false><<<LNB, 256, 0, stream>>>(X, nullptr, blg, blb, Ab);
        gemm_bitlinear_kernel<<<ggrid, 256, 0, stream>>>(Ab, Wb, blbias, Yb, s0);
        dequant_kernel<false, false, u16><<<DQB, 256, 0, stream>>>(Yb, s0, wn, 0, nullptr, Ab);
        flash_attn_kernel<<<fgrid, 128, 0, stream>>>(Ab, Tb);

        // x = bitlinear(attn_out + x)
        ln_kernel<u16, true><<<LNB, 256, 0, stream>>>(Tb, X, blg, blb, Ab);
        gemm_bitlinear_kernel<<<ggrid, 256, 0, stream>>>(Ab, Wb, blbias, Yb, s1);
        dequant_kernel<false, false, float><<<DQB, 256, 0, stream>>>(Yb, s1, wn, 0, nullptr, X);

        // FF: LN -> bitlinear(f1) -> gelu -> bitlinear(f2) ; residual
        ln_kernel<float, false><<<LNB, 256, 0, stream>>>(X, nullptr, ffg + i * 1024, ffb + i * 1024, Tb);
        ln_kernel<u16, false><<<LNB, 256, 0, stream>>>(Tb, nullptr, f1g + i * 1024, f1b + i * 1024, Ab);
        gemm_bitlinear_kernel<<<ggrid, 256, 0, stream>>>(Ab, Wb + (size_t)(1 + i) * WSZ, f1bb + i * 1024, Yb, s2);
        dequant_kernel<true, false, float><<<DQB, 256, 0, stream>>>(Yb, s2, wn, 1 + i, nullptr, Tb);
        ln_kernel<u16, false><<<LNB, 256, 0, stream>>>(Tb, nullptr, f2g + i * 1024, f2b + i * 1024, Ab);
        gemm_bitlinear_kernel<<<ggrid, 256, 0, stream>>>(Ab, Wb + (size_t)(7 + i) * WSZ, f2bb + i * 1024, Yb, s3);
        dequant_kernel<false, true, float><<<DQB, 256, 0, stream>>>(Yb, s3, wn, 7 + i, X, X);
    }

    ln_kernel<float, false><<<LNB, 256, 0, stream>>>(X, nullptr, ng, nb, (float*)d_out);
}
